// AttnDecoderLSTM_52604759441991
// MI455X (gfx1250) — compile-verified, matrix operand order checked by reading, not run on hardware
//
#include <hip/hip_runtime.h>
#include <math.h>

typedef __attribute__((ext_vector_type(16))) _Float16 v16h;
typedef __attribute__((ext_vector_type(8)))  _Float16 h8;
typedef __attribute__((ext_vector_type(8)))  float    v8f;
typedef __attribute__((ext_vector_type(4)))  float    f4;
typedef int v4ig __attribute__((vector_size(16)));   // matches builtin param type

#define BATCH   512
#define SEQ     256
#define EMB     256
#define FEAT    512
#define HID     1024
#define KXH     1792      // EMB + FEAT + HID
#define G4      4096      // 4*HID
#define NOUT    14

#define AS1 __attribute__((address_space(1)))
#define AS3 __attribute__((address_space(3)))

// CDNA5 async global->LDS staging if the toolchain exposes it; otherwise fall
// back to the global_load_b128 + ds_store_b128 path.
#if defined(__has_builtin)
#  if __has_builtin(__builtin_amdgcn_global_load_async_to_lds_b128) && \
      __has_builtin(__builtin_amdgcn_s_wait_asynccnt)
#    define USE_ASYNC_LDS 1
#  else
#    define USE_ASYNC_LDS 0
#  endif
#else
#  define USE_ASYNC_LDS 0
#endif

#if USE_ASYNC_LDS
// generic LDS pointer low 32 bits == LDS byte offset (ISA 10.2); flat address
// of global memory == global address.
__device__ __forceinline__ AS3 v4ig* lds_cast(void* p) {
    return (AS3 v4ig*)(unsigned int)(unsigned long long)p;
}
__device__ __forceinline__ AS1 v4ig* gbl_cast(const void* p) {
    return (AS1 v4ig*)(unsigned long long)p;
}
#endif

// ---------------------------------------------------------------------------
// f32 -> f16 2D convert with independent pitches (packs W_ih|W_hh into one
// K-major matrix, converts W_in / W_out).
// ---------------------------------------------------------------------------
__global__ void cvt2d_f16(_Float16* __restrict__ dst, int dpitch, int dcol0,
                          const float* __restrict__ src, int spitch,
                          int rows, int cols) {
    int total = rows * cols;
    for (int i = blockIdx.x * blockDim.x + threadIdx.x; i < total;
         i += gridDim.x * blockDim.x) {
        int r = i / cols, c = i - r * cols;
        dst[r * dpitch + dcol0 + c] = (_Float16)src[r * spitch + c];
    }
}

// Pack x = [embedding[action] | feature | h0] as f16 [B, 1792]
__global__ void pack_xh(const int* __restrict__ action,
                        const float* __restrict__ feat,
                        const float* __restrict__ h0,
                        const float* __restrict__ emb,
                        _Float16* __restrict__ xh) {
    int total = BATCH * KXH;
    for (int i = blockIdx.x * blockDim.x + threadIdx.x; i < total;
         i += gridDim.x * blockDim.x) {
        int b = i / KXH, c = i - b * KXH;
        float v;
        if (c < EMB)             v = emb[action[b] * EMB + c];
        else if (c < EMB + FEAT) v = feat[b * FEAT + (c - EMB)];
        else                     v = h0[b * HID + (c - EMB - FEAT)];
        xh[i] = (_Float16)v;
    }
}

__global__ void bias_sum(const float* __restrict__ a, const float* __restrict__ b,
                         float* __restrict__ o) {
    int i = blockIdx.x * blockDim.x + threadIdx.x;
    if (i < G4) o[i] = a[i] + b[i];
}

// ---------------------------------------------------------------------------
// WMMA GEMM: C[M,N] = A[M,K] * Bw[N,K]^T (+bias[n]).  A,Bw f16 row-major,
// C f32.  Block tile 64x128: 8 waves, each wave owns a 32x32 tile (2x2 WMMA).
// Double-buffered LDS staging, async global->LDS when available.
// M%64==0, N%128==0, K%32==0 (true for all calls here).
// ---------------------------------------------------------------------------
__launch_bounds__(256, 2)
__global__ void gemm_f16_tn(const _Float16* __restrict__ A,
                            const _Float16* __restrict__ Bw,
                            float* __restrict__ C,
                            const float* __restrict__ bias,
                            int M, int N, int K) {
    __shared__ __align__(16) _Float16 As[2][64][32];
    __shared__ __align__(16) _Float16 Bs[2][128][32];

    const int tid  = threadIdx.x;
    const int lane = tid & 31;
    const int w    = tid >> 5;      // 0..7
    const int wm   = w >> 2;        // 0..1  (M direction)
    const int wn   = w & 3;         // 0..3  (N direction)

    const int nbn = N >> 7;
    const int bm  = blockIdx.x / nbn;
    const int bn  = blockIdx.x - bm * nbn;
    const int m0  = bm << 6;
    const int n0  = bn << 7;

    // cooperative staging indices
    const int arow = tid >> 2;            // 0..63
    const int acol = (tid & 3) * 8;       // 0,8,16,24
    const int brow = tid >> 1;            // 0..127
    const int bcol = (tid & 1) * 16;      // 0,16

    const int rsel = lane & 15;           // fragment row/col select
    const int g    = lane >> 4;           // K-chunk group

    const _Float16* aSrc = &A[(m0 + arow) * K + acol];
    const _Float16* bSrc = &Bw[(n0 + brow) * K + bcol];

    v8f acc[2][2] = {};

    auto stage = [&](int buf, int k0) {
#if USE_ASYNC_LDS
        __builtin_amdgcn_global_load_async_to_lds_b128(
            gbl_cast(aSrc + k0), lds_cast(&As[buf][arow][acol]), 0, 0);
        __builtin_amdgcn_global_load_async_to_lds_b128(
            gbl_cast(bSrc + k0), lds_cast(&Bs[buf][brow][bcol]), 0, 0);
        __builtin_amdgcn_global_load_async_to_lds_b128(
            gbl_cast(bSrc + k0 + 8), lds_cast(&Bs[buf][brow][bcol + 8]), 0, 0);
#else
        *(h8*)&As[buf][arow][acol]   = *(const h8*)(aSrc + k0);
        *(h8*)&Bs[buf][brow][bcol]   = *(const h8*)(bSrc + k0);
        *(h8*)&Bs[buf][brow][bcol+8] = *(const h8*)(bSrc + k0 + 8);
#endif
    };

    stage(0, 0);

    for (int k0 = 0; k0 < K; k0 += 32) {
        const int cur = (k0 >> 5) & 1;
        if (k0 + 32 < K) {
            stage(cur ^ 1, k0 + 32);          // pipeline next slab
#if USE_ASYNC_LDS
            __builtin_amdgcn_s_wait_asynccnt(3);   // current slab resident
#endif
        } else {
#if USE_ASYNC_LDS
            __builtin_amdgcn_s_wait_asynccnt(0);
#endif
        }
        if (k0 + 64 < K) {                    // warm L2/WGP$ two slabs ahead
            __builtin_prefetch(aSrc + k0 + 64, 0, 1);
            __builtin_prefetch(bSrc + k0 + 64, 0, 1);
        }
        __syncthreads();

        v16h af[2], bf[2];
#pragma unroll
        for (int i = 0; i < 2; ++i) {
            int ar = wm * 32 + i * 16 + rsel;
            h8 lo = *(const h8*)&As[cur][ar][8 * g];
            h8 hi = *(const h8*)&As[cur][ar][16 + 8 * g];
            af[i] = __builtin_shufflevector(lo, hi, 0,1,2,3,4,5,6,7,8,9,10,11,12,13,14,15);
        }
#pragma unroll
        for (int j = 0; j < 2; ++j) {
            int br = wn * 32 + j * 16 + rsel;
            h8 lo = *(const h8*)&Bs[cur][br][8 * g];
            h8 hi = *(const h8*)&Bs[cur][br][16 + 8 * g];
            bf[j] = __builtin_shufflevector(lo, hi, 0,1,2,3,4,5,6,7,8,9,10,11,12,13,14,15);
        }
#pragma unroll
        for (int i = 0; i < 2; ++i)
#pragma unroll
            for (int j = 0; j < 2; ++j)
                acc[i][j] = __builtin_amdgcn_wmma_f32_16x16x32_f16(
                    false, af[i], false, bf[j], (short)0, acc[i][j], false, false);
        __syncthreads();
    }

    // C/D layout: lane l -> N = l&15; VGPR v -> M = v + 8*(l>>4)
    const int cn = lane & 15;
    const int cg = lane >> 4;
#pragma unroll
    for (int i = 0; i < 2; ++i)
#pragma unroll
        for (int j = 0; j < 2; ++j) {
            int nIdx = n0 + wn * 32 + j * 16 + cn;
            float bv = bias ? bias[nIdx] : 0.0f;
#pragma unroll
            for (int v = 0; v < 8; ++v) {
                int mIdx = m0 + wm * 32 + i * 16 + v + 8 * cg;
                C[mIdx * N + nIdx] = acc[i][j][v] + bv;
            }
        }
}

// ---------------------------------------------------------------------------
// LSTM cell elementwise: gates [B,4H] order i,f,g,o
// ---------------------------------------------------------------------------
__device__ __forceinline__ float sigm(float x) { return 1.0f / (1.0f + expf(-x)); }

__global__ void lstm_cell(const float* __restrict__ gates,
                          const float* __restrict__ c0,
                          float* __restrict__ h1_out, float* __restrict__ c1_out,
                          _Float16* __restrict__ h1_h, _Float16* __restrict__ wh_h) {
    int idx = blockIdx.x * blockDim.x + threadIdx.x;
    if (idx >= BATCH * HID) return;
    int b = idx >> 10, j = idx & (HID - 1);
    const float* gr = gates + b * G4;
    float ig = sigm(gr[j]);
    float fg = sigm(gr[HID + j]);
    float gg = tanhf(gr[2 * HID + j]);
    float og = sigm(gr[3 * HID + j]);
    float c1 = fg * c0[idx] + ig * gg;
    float h1 = og * tanhf(c1);
    h1_out[idx] = h1;
    c1_out[idx] = c1;
    h1_h[idx] = (_Float16)h1;
    wh_h[b * (2 * HID) + HID + j] = (_Float16)h1;   // second half of [weighted|h1]
}

// ---------------------------------------------------------------------------
// Per-batch attention: scores vs ctx, masked softmax, weighted ctx sum.
// One block (256 thr = 8 waves) per batch element; ctx stays f32 (HBM-bound).
// ---------------------------------------------------------------------------
__launch_bounds__(256)
__global__ void attn_kernel(const float* __restrict__ ctx,
                            const unsigned char* __restrict__ mask,
                            const float* __restrict__ target,
                            float* __restrict__ alpha_out,
                            _Float16* __restrict__ wh_h) {
    const int b = blockIdx.x;
    __shared__ float t[HID];
    __shared__ float attn[SEQ];
    __shared__ float red[8];
    __shared__ float smax, ssum;

    const int tid = threadIdx.x;
    const int lane = tid & 31, w = tid >> 5;
    const float* cb = ctx + (size_t)b * SEQ * HID;

    for (int d = tid; d < HID; d += 256) t[d] = target[b * HID + d];
    __syncthreads();

    // scores: wave w handles rows s = w, w+8, ... (lane-parallel dot, coalesced)
    for (int s = w; s < SEQ; s += 8) {
        const float* row = cb + s * HID;
        float sum = 0.0f;
        for (int d = lane * 4; d < HID; d += 128) {
            f4 v = *(const f4*)&row[d];
            sum += v.x * t[d] + v.y * t[d + 1] + v.z * t[d + 2] + v.w * t[d + 3];
        }
#pragma unroll
        for (int off = 16; off > 0; off >>= 1) sum += __shfl_xor(sum, off, 32);
        if (lane == 0) attn[s] = mask[b * SEQ + s] ? -INFINITY : sum;
    }
    __syncthreads();

    // softmax over 256
    float v = attn[tid];
    float m = v;
#pragma unroll
    for (int off = 16; off > 0; off >>= 1) m = fmaxf(m, __shfl_xor(m, off, 32));
    if (lane == 0) red[w] = m;
    __syncthreads();
    if (tid == 0) {
        float mm = red[0];
        for (int i = 1; i < 8; ++i) mm = fmaxf(mm, red[i]);
        smax = mm;
    }
    __syncthreads();
    float e = expf(v - smax);
    float p = e;
#pragma unroll
    for (int off = 16; off > 0; off >>= 1) p += __shfl_xor(p, off, 32);
    if (lane == 0) red[w] = p;
    __syncthreads();
    if (tid == 0) {
        float ss = red[0];
        for (int i = 1; i < 8; ++i) ss += red[i];
        ssum = ss;
    }
    __syncthreads();
    float a = e / ssum;
    attn[tid] = a;
    alpha_out[b * SEQ + tid] = a;
    __syncthreads();

    // weighted[d] = sum_s alpha[s] * ctx[b,s,d]  (coalesced across d).
    // Last use of ctx[b] -> non-temporal so we don't displace other blocks.
    int d0 = tid * 4;
    float a0 = 0.f, a1 = 0.f, a2 = 0.f, a3 = 0.f;
    for (int s = 0; s < SEQ; ++s) {
        float al = attn[s];
        f4 c = __builtin_nontemporal_load((const f4*)&cb[s * HID + d0]);
        a0 += al * c.x; a1 += al * c.y; a2 += al * c.z; a3 += al * c.w;
    }
    _Float16* whb = wh_h + b * (2 * HID);   // first half of [weighted|h1]
    whb[d0]     = (_Float16)a0;
    whb[d0 + 1] = (_Float16)a1;
    whb[d0 + 2] = (_Float16)a2;
    whb[d0 + 3] = (_Float16)a3;
}

__global__ void htilde_kernel(const float* __restrict__ pre, _Float16* __restrict__ ht) {
    int i = blockIdx.x * blockDim.x + threadIdx.x;
    if (i < BATCH * HID) ht[i] = (_Float16)tanhf(pre[i]);
}

__global__ void logit_kernel(const _Float16* __restrict__ ht,
                             const float* __restrict__ Wdec,
                             const float* __restrict__ bdec,
                             float* __restrict__ out) {
    int idx = blockIdx.x * blockDim.x + threadIdx.x;
    if (idx >= BATCH * NOUT) return;
    int b = idx / NOUT, o = idx - b * NOUT;
    const _Float16* h = ht + b * HID;
    const float* wr = Wdec + o * HID;
    float s = 0.0f;
    for (int d = 0; d < HID; ++d) s += (float)h[d] * wr[d];
    out[idx] = s + bdec[o];
}

// ---------------------------------------------------------------------------
extern "C" void kernel_launch(void* const* d_in, const int* in_sizes, int n_in,
                              void* d_out, int out_size, void* d_ws, size_t ws_size,
                              hipStream_t stream) {
    const int*   action  = (const int*)d_in[0];
    const float* feature = (const float*)d_in[1];
    const float* h0      = (const float*)d_in[2];
    const float* c0      = (const float*)d_in[3];
    const float* ctx     = (const float*)d_in[4];
    const unsigned char* mask = (const unsigned char*)d_in[5];
    const float* emb     = (const float*)d_in[6];
    const float* Wih     = (const float*)d_in[7];
    const float* Whh     = (const float*)d_in[8];
    const float* bih     = (const float*)d_in[9];
    const float* bhh     = (const float*)d_in[10];
    const float* Win     = (const float*)d_in[11];
    const float* Wout    = (const float*)d_in[12];
    const float* Wdec    = (const float*)d_in[13];
    const float* bdec    = (const float*)d_in[14];

    // workspace carve (all offsets 16B-aligned); total ~37.5 MB
    char* ws = (char*)d_ws;
    _Float16* Wg_h   = (_Float16*)(ws + 0);          // [4096][1792]
    _Float16* Win_h  = (_Float16*)(ws + 14680064);   // [1024][1024]
    _Float16* Wout_h = (_Float16*)(ws + 16777216);   // [1024][2048]
    _Float16* xh_h   = (_Float16*)(ws + 20971520);   // [512][1792]
    _Float16* h1_h   = (_Float16*)(ws + 22806528);   // [512][1024]
    _Float16* wh_h   = (_Float16*)(ws + 23855104);   // [512][2048]
    _Float16* ht_h   = (_Float16*)(ws + 25952256);   // [512][1024]
    float*    bsum   = (float*)   (ws + 27000832);   // [4096]
    float*    gates  = (float*)   (ws + 27017216);   // [512][4096] (reused as pre-act)
    float*    target = (float*)   (ws + 35405824);   // [512][1024]

    float* out       = (float*)d_out;
    float* h1_out    = out;                 // 512*1024
    float* c1_out    = out + 524288;        // 512*1024
    float* alpha_out = out + 1048576;       // 512*256
    float* logit_out = out + 1179648;       // 512*14

    // 1. pack / convert to f16
    cvt2d_f16<<<(G4 * 768 + 255) / 256, 256, 0, stream>>>(Wg_h, KXH, 0, Wih, EMB + FEAT, G4, 768);
    cvt2d_f16<<<(G4 * HID + 255) / 256, 256, 0, stream>>>(Wg_h, KXH, 768, Whh, HID, G4, HID);
    cvt2d_f16<<<(HID * HID + 255) / 256, 256, 0, stream>>>(Win_h, HID, 0, Win, HID, HID, HID);
    cvt2d_f16<<<(HID * 2 * HID + 255) / 256, 256, 0, stream>>>(Wout_h, 2 * HID, 0, Wout, 2 * HID, HID, 2 * HID);
    pack_xh<<<(BATCH * KXH + 255) / 256, 256, 0, stream>>>(action, feature, h0, emb, xh_h);
    bias_sum<<<G4 / 256, 256, 0, stream>>>(bih, bhh, bsum);

    // 2. gates = [x|h0] @ [Wih|Whh]^T + (bih+bhh)   [512,4096]
    gemm_f16_tn<<<(BATCH / 64) * (G4 / 128), 256, 0, stream>>>(xh_h, Wg_h, gates, bsum, BATCH, G4, KXH);

    // 3. LSTM cell -> h1, c1
    lstm_cell<<<(BATCH * HID + 255) / 256, 256, 0, stream>>>(gates, c0, h1_out, c1_out, h1_h, wh_h);

    // 4. target = h1 @ Win^T   [512,1024]
    gemm_f16_tn<<<(BATCH / 64) * (HID / 128), 256, 0, stream>>>(h1_h, Win_h, target, nullptr, BATCH, HID, HID);

    // 5. attention: scores, masked softmax (alpha out), weighted ctx -> wh_h
    attn_kernel<<<BATCH, 256, 0, stream>>>(ctx, mask, target, alpha_out, wh_h);

    // 6. pre = [weighted|h1] @ Wout^T  (reuse gates buffer)
    gemm_f16_tn<<<(BATCH / 64) * (HID / 128), 256, 0, stream>>>(wh_h, Wout_h, gates, nullptr, BATCH, HID, 2 * HID);

    // 7. h_tilde = tanh(pre) -> f16
    htilde_kernel<<<(BATCH * HID + 255) / 256, 256, 0, stream>>>(gates, ht_h);

    // 8. logits = h_tilde @ Wdec^T + bdec
    logit_kernel<<<(BATCH * NOUT + 255) / 256, 256, 0, stream>>>(ht_h, Wdec, bdec, logit_out);
}
